// LSTMLearner_36593121362382
// MI455X (gfx1250) — matrix operand order repaired and checked by reading, not compiled, not run on hardware
//
#include <hip/hip_runtime.h>
#include <math.h>

#define B_    32
#define TSEQ  64
#define TCAP  63
#define VOCAB 32000
#define IMGD  20480
#define EMB   512
#define H_    500
#define G4    2000   // 4*H (original layout)
#define G4P   2048   // 4*512 slab-padded gate layout
#define KP    512    // K padded to multiple of 32 for bf16 WMMA
#define NT_V  (VOCAB / 16)   // 2000 n-tiles in the logits GEMM
#define NB_STEP 4            // blocks in the persistent recurrence kernel

typedef __bf16 bf16_t;
typedef __attribute__((ext_vector_type(16))) __bf16 v16bf;
typedef __attribute__((ext_vector_type(8)))  __bf16 v8bf;
typedef __attribute__((ext_vector_type(8)))  float  v8f;

__device__ __forceinline__ unsigned short f2bf(float f) {
  unsigned u = __float_as_uint(f);
  unsigned r = u + 0x7FFFu + ((u >> 16) & 1u);   // round-to-nearest-even
  return (unsigned short)(r >> 16);
}
__device__ __forceinline__ float sigm(float x) { return 1.f / (1.f + __expf(-x)); }
__device__ __forceinline__ float tanh_fast(float x) {
  float ax = fabsf(x);
  float e = __expf(-2.f * ax);
  float r = (1.f - e) / (1.f + e);
  return copysignf(r, x);
}
__device__ __forceinline__ v16bf load_afrag(const bf16_t* ap) {
  // A frag (ISA 7.12.2, 16-bit A 16x32): lane<16 -> K {0..7,16..23},
  // lane>=16 -> K {8..15,24..31}; two 16B chunks 32B apart.
  v8bf alo = *(const v8bf*)ap;
  v8bf ahi = *(const v8bf*)(ap + 16);
  return __builtin_shufflevector(alo, ahi,
      0, 1, 2, 3, 4, 5, 6, 7, 8, 9, 10, 11, 12, 13, 14, 15);
}

// Device-wide barrier: monotonic counter, no reset => no re-arm race.
// Deterministic (spin order does not affect arithmetic).
__device__ __forceinline__ void global_barrier(int* cnt, int target) {
  __syncthreads();
  __threadfence();                       // release: make h writes visible
  if (threadIdx.x == 0) {
    atomicAdd(cnt, 1);
    while (atomicAdd(cnt, 0) < target) { __builtin_amdgcn_s_sleep(2); }
  }
  __syncthreads();
  __threadfence();                       // acquire: invalidate stale lines
}

// ---------------- utility kernels ----------------

__global__ void zero_ws_kernel(uint4* p, long n16) {
  long i = (long)blockIdx.x * blockDim.x + threadIdx.x;
  if (i < n16) p[i] = make_uint4(0u, 0u, 0u, 0u);
}

// x[b*TSEQ + 0, :] = img[b,:] @ W_img + b_img   (M=32, K=20480, N=512)
__global__ void img_embed_kernel(const float* __restrict__ img,
                                 const float* __restrict__ W,
                                 const float* __restrict__ bias,
                                 float* __restrict__ x) {
  int b = blockIdx.x;
  int j = threadIdx.x * 4;                 // 128 threads * 4 cols = 512
  float4 acc = *(const float4*)(bias + j);
  const float* irow = img + (size_t)b * IMGD;
  const float* wp = W + j;
  for (int k = 0; k < IMGD; ++k) {
    float iv = irow[k];
    float4 w = *(const float4*)(wp + (size_t)k * EMB);
    acc.x = fmaf(iv, w.x, acc.x);
    acc.y = fmaf(iv, w.y, acc.y);
    acc.z = fmaf(iv, w.z, acc.z);
    acc.w = fmaf(iv, w.w, acc.w);
  }
  *(float4*)(x + (size_t)b * TSEQ * EMB + j) = acc;
}

// x[b*TSEQ + 1 + t, :] = emb[cap[b,t], :]
__global__ void gather_kernel(const int* __restrict__ cap,
                              const float* __restrict__ emb,
                              float* __restrict__ x) {
  int id = blockIdx.x;                    // b*TCAP + t
  int b = id / TCAP, t = id - b * TCAP;
  int tok = cap[id];
  const float4* s = (const float4*)(emb + (size_t)tok * EMB);
  float4* d = (float4*)(x + (size_t)(b * TSEQ + 1 + t) * EMB);
  d[threadIdx.x] = s[threadIdx.x];        // 128 threads * float4 = 512
}

__global__ void cvt_bf16_kernel(const float* __restrict__ src,
                                unsigned short* __restrict__ dst, int n) {
  int i = blockIdx.x * blockDim.x + threadIdx.x;
  if (i < n) dst[i] = f2bf(src[i]);
}

// dst[k, n] = k < Ks ? bf16(src[k, n]) : 0   -- pads K=500 -> 512 (W_out)
__global__ void cvt_pad_kernel(const float* __restrict__ src,
                               unsigned short* __restrict__ dst,
                               int Ks, int N, long total) {
  long i = (long)blockIdx.x * blockDim.x + threadIdx.x;
  if (i >= total) return;
  long k = i / N;
  long n = i - k * N;
  dst[i] = (k < Ks) ? f2bf(src[k * (long)N + n]) : (unsigned short)0;
}

// src [Ksrc x 2000] (gate slabs of 500) -> dst bf16 [512 x 2048] (slabs of 512,
// zero padded in both k and u). Used for W_lstm (Ks=512) and U (Ks=500).
__global__ void repack_pad_kernel(const float* __restrict__ src,
                                  unsigned short* __restrict__ dst,
                                  int Ks, long total) {
  long i = (long)blockIdx.x * blockDim.x + threadIdx.x;
  if (i >= total) return;
  int k = (int)(i >> 11);          // / 2048
  int col = (int)(i & 2047);
  int g = col >> 9, u = col & 511;
  dst[i] = (k < Ks && u < H_) ? f2bf(src[(size_t)k * G4 + g * H_ + u])
                              : (unsigned short)0;
}

// b_lstm [2000] -> [2048] slab-padded fp32
__global__ void repack_bias_kernel(const float* __restrict__ src,
                                   float* __restrict__ dst) {
  int col = blockIdx.x * blockDim.x + threadIdx.x;
  if (col >= G4P) return;
  int g = col >> 9, u = col & 511;
  dst[col] = (u < H_) ? src[g * H_ + u] : 0.f;
}

// ---------------- bf16 WMMA GEMM ----------------
// C[M,N] = A_bf16[M,K] @ B_bf16[K,N] (+ bias[N]).
// One wave owns a (16*MT) x 16 tile; 8 waves per block share the same M rows.
// PSTATS: emit per-(row, n-tile) softmax partials {rowmax, sum exp(x-max)}
// via 16-lane xor-shuffle reductions (stay within each half-wave => match the
// C layout where a row's 16 columns live in lanes 0-15 or 16-31).
template <int MT, bool PSTATS>
__global__ __launch_bounds__(256) void wmma_gemm_bf16(
    const bf16_t* __restrict__ A, int lda,
    const bf16_t* __restrict__ Bm, int ldb,
    const float* __restrict__ bias,
    float* __restrict__ C, int ldc,
    float2* __restrict__ pstats, int ntstride,
    int N, int K) {
  const int lane = threadIdx.x & 31;
  const int wave = threadIdx.x >> 5;
  const int nt = blockIdx.x * 8 + wave;
  if (nt * 16 >= N) return;
  const int n0 = nt * 16;
  const int m0 = blockIdx.y * (16 * MT);
  const int half = lane >> 4;     // 0 for lanes 0-15, 1 for 16-31
  const int l15 = lane & 15;

  v8f acc[MT] = {};

  for (int k0 = 0; k0 < K; k0 += 32) {
    // B frag: lane holds column N=n0+l15, K {8h..8h+7, 16+8h..16+8h+7}
    const bf16_t* bp = Bm + (size_t)(k0 + half * 8) * ldb + n0 + l15;
    v16bf bfr;
#pragma unroll
    for (int s = 0; s < 16; ++s) bfr[s] = bp[(size_t)((s < 8) ? s : s + 8) * ldb];
#pragma unroll
    for (int mi = 0; mi < MT; ++mi) {
      v16bf afr = load_afrag(A + (size_t)(m0 + mi * 16 + l15) * lda + k0 + half * 8);
      acc[mi] = __builtin_amdgcn_wmma_f32_16x16x32_bf16(
          false, afr, false, bfr, (short)0, acc[mi], false, false);
    }
  }

  const float bv = bias ? bias[n0 + l15] : 0.f;
#pragma unroll
  for (int mi = 0; mi < MT; ++mi) {
#pragma unroll
    for (int v = 0; v < 8; ++v) {
      int row = m0 + mi * 16 + v + half * 8;   // C/D layout: VGPR v -> M=v(+8)
      float r = acc[mi][v] + bv;
      C[(size_t)row * ldc + n0 + l15] = r;
      if (PSTATS) {
        float mx = r;
#pragma unroll
        for (int msk = 1; msk <= 8; msk <<= 1)
          mx = fmaxf(mx, __shfl_xor(mx, msk, 32));
        float se = __expf(r - mx);
#pragma unroll
        for (int msk = 1; msk <= 8; msk <<= 1)
          se += __shfl_xor(se, msk, 32);
        if (l15 == 0)
          pstats[(size_t)row * ntstride + nt] = make_float2(mx, se);
      }
    }
  }
}

// ---------------- persistent LSTM recurrence: ONE launch for all 64 steps ---
// 4 blocks x 8 waves; wave owns u-tile ut across all 4 gate slabs.
// Per step: z = h_prev @ U + xz[t]; gates; h,c update. Cell state c lives in
// registers (same lane owns the same (b,u) all 64 steps). h ping-pongs in
// global, separated by a device-wide barrier (release/acquire fences).
__global__ __launch_bounds__(256) void lstm_persist_kernel(
    const bf16_t* __restrict__ Ubf,        // [512][2048] slab-padded
    const float* __restrict__ xz,          // [2048][2048], row r=b*TSEQ+t
    unsigned short* __restrict__ h0,       // [32][512] bf16 (zeroed)
    unsigned short* __restrict__ h1,       // [32][512] bf16 (zeroed)
    unsigned short* __restrict__ h_all_bf, // [2048][512] bf16
    int* __restrict__ bar_cnt) {
  const int lane = threadIdx.x & 31;
  const int wave = threadIdx.x >> 5;
  const int ut = blockIdx.x * 8 + wave;    // 0..31 (u-tile)
  const int u0 = ut * 16;
  const int half = lane >> 4;
  const int l15 = lane & 15;
  const int u = u0 + l15;
  const bool valid = (u < H_);

  float creg[2][8] = {};                   // cell state, register-resident

  for (int t = 0; t < TSEQ; ++t) {
    const bf16_t* h_src = (const bf16_t*)((t & 1) ? h1 : h0);
    unsigned short* h_dst = (t & 1) ? h0 : h1;

    v8f acc[4][2] = {};
    for (int k0 = 0; k0 < KP; k0 += 32) {
      v16bf afr[2];
#pragma unroll
      for (int mi = 0; mi < 2; ++mi)
        afr[mi] = load_afrag(h_src + (size_t)(mi * 16 + l15) * KP + k0 + half * 8);
#pragma unroll
      for (int g = 0; g < 4; ++g) {
        const bf16_t* bp = Ubf + (size_t)(k0 + half * 8) * G4P + g * 512 + u0 + l15;
        v16bf bfr;   // column u0+l15, K {8h..8h+7, 16+8h..16+8h+7}
#pragma unroll
        for (int s = 0; s < 16; ++s) bfr[s] = bp[(size_t)((s < 8) ? s : s + 8) * G4P];
#pragma unroll
        for (int mi = 0; mi < 2; ++mi)
          acc[g][mi] = __builtin_amdgcn_wmma_f32_16x16x32_bf16(
              false, afr[mi], false, bfr, (short)0, acc[g][mi], false, false);
      }
    }

#pragma unroll
    for (int mi = 0; mi < 2; ++mi) {
#pragma unroll
      for (int v = 0; v < 8; ++v) {
        int b = mi * 16 + v + half * 8;               // batch row 0..31
        if (valid) {
          size_t xr = (size_t)(b * TSEQ + t) * G4P;
          float zi = acc[0][mi][v] + xz[xr + 0 * 512 + u];
          float zf = acc[1][mi][v] + xz[xr + 1 * 512 + u];
          float zg = acc[2][mi][v] + xz[xr + 2 * 512 + u];
          float zo = acc[3][mi][v] + xz[xr + 3 * 512 + u];
          float gi = sigm(zi), gf = sigm(zf);
          float gg = tanh_fast(zg), go = sigm(zo);
          float cn = gf * creg[mi][v] + gi * gg;
          creg[mi][v] = cn;
          unsigned short hb = f2bf(go * tanh_fast(cn));
          h_dst[b * KP + u] = hb;                     // cols 500..511 stay 0
          h_all_bf[(size_t)(b * TSEQ + t) * KP + u] = hb;
        }
      }
    }
    // all blocks must see the new h before step t+1 reads it
    global_barrier(bar_cnt, NB_STEP * (t + 1));
  }
}

// ---------------- softmax stats reduce (32 MB instead of 256 MB pass) -------
__global__ void stats_reduce_kernel(const float2* __restrict__ pstats,
                                    float* __restrict__ stats) {
  int r = blockIdx.x;
  const float2* pr = pstats + (size_t)r * NT_V;
  float m = -INFINITY, s = 0.f;
  for (int i = threadIdx.x; i < NT_V; i += blockDim.x) {
    float2 p = pr[i];
    float mn = fmaxf(m, p.x);
    s = s * __expf(m - mn) + p.y * __expf(p.x - mn);
    m = mn;
  }
  __shared__ float sm[256], ss[256];
  int tid = threadIdx.x;
  sm[tid] = m; ss[tid] = s;
  __syncthreads();
  for (int off = 128; off > 0; off >>= 1) {
    if (tid < off) {
      float m2 = sm[tid + off], s2 = ss[tid + off];
      float m1 = sm[tid],       s1 = ss[tid];
      float mn = fmaxf(m1, m2);
      ss[tid] = s1 * __expf(m1 - mn) + s2 * __expf(m2 - mn);
      sm[tid] = mn;
    }
    __syncthreads();
  }
  if (tid == 0) { stats[2 * r] = sm[0]; stats[2 * r + 1] = ss[0]; }
}

__global__ void softmax_norm_kernel(float* __restrict__ logits,
                                    const float* __restrict__ stats) {
  int r = blockIdx.y;
  int i = blockIdx.x * blockDim.x + threadIdx.x;
  if (i >= VOCAB) return;
  float m = stats[2 * r];
  float inv = 1.f / stats[2 * r + 1];
  size_t o = (size_t)r * VOCAB + i;
  // final output is never re-read: non-temporal store keeps L2 for weights
  __builtin_nontemporal_store(__expf(logits[o] - m) * inv, &logits[o]);
}

// ---------------- launch ----------------
extern "C" void kernel_launch(void* const* d_in, const int* in_sizes, int n_in,
                              void* d_out, int out_size, void* d_ws, size_t ws_size,
                              hipStream_t stream) {
  const float* img    = (const float*)d_in[0];
  const int*   cap    = (const int*)d_in[1];
  const float* W_img  = (const float*)d_in[2];
  const float* b_img  = (const float*)d_in[3];
  const float* emb    = (const float*)d_in[4];
  const float* W_lstm = (const float*)d_in[5];
  const float* U_lstm = (const float*)d_in[6];
  const float* b_lstm = (const float*)d_in[7];
  const float* W_out  = (const float*)d_in[8];
  const float* b_out  = (const float*)d_in[9];
  float* out = (float*)d_out;
  char* ws = (char*)d_ws;

  size_t off = 0;
  auto take = [&](size_t bytes) { size_t o = off; off += (bytes + 255) & ~(size_t)255; return o; };
  // zeroed region first & contiguous: h ping-pong pair, h_all, barrier counter
  size_t o_h0    = take((size_t)B_ * KP * 2);
  size_t o_h1    = take((size_t)B_ * KP * 2);
  size_t o_hall  = take((size_t)B_ * TSEQ * KP * 2);
  size_t o_bar   = take(256);
  size_t zero_bytes = off;
  size_t o_x     = take((size_t)B_ * TSEQ * EMB * 4);
  size_t o_xbf   = take((size_t)B_ * TSEQ * EMB * 2);
  size_t o_wl    = take((size_t)KP * G4P * 2);          // slab-padded W_lstm
  size_t o_ubf   = take((size_t)KP * G4P * 2);          // slab-padded U
  size_t o_bp    = take((size_t)G4P * 4);               // slab-padded b_lstm
  size_t o_wo    = take((size_t)KP * VOCAB * 2);
  size_t o_xz    = take((size_t)B_ * TSEQ * G4P * 4);   // [2048][2048]
  size_t o_pst   = take((size_t)B_ * TSEQ * NT_V * 8);  // float2 partials, 32 MB
  size_t o_stats = take((size_t)B_ * TSEQ * 2 * 4);
  (void)ws_size; (void)in_sizes; (void)n_in; (void)out_size;

  // 0) zero h ping-pong / h_all / barrier counter (h padding cols must be 0)
  {
    long n16 = (long)(zero_bytes / 16);
    zero_ws_kernel<<<(n16 + 255) / 256, 256, 0, stream>>>((uint4*)ws, n16);
  }
  // 1) x[:,0,:] = image embed ; x[:,1:,:] = caption gather
  img_embed_kernel<<<B_, 128, 0, stream>>>(img, W_img, b_img, (float*)(ws + o_x));
  gather_kernel<<<B_ * TCAP, 128, 0, stream>>>(cap, emb, (float*)(ws + o_x));
  // 2) bf16 conversions / slab repacks (weights L2-resident afterwards)
  { int n = B_ * TSEQ * EMB;
    cvt_bf16_kernel<<<(n + 255) / 256, 256, 0, stream>>>((const float*)(ws + o_x),
                                                         (unsigned short*)(ws + o_xbf), n); }
  { long n = (long)KP * G4P;
    repack_pad_kernel<<<(n + 255) / 256, 256, 0, stream>>>(W_lstm,
                                                        (unsigned short*)(ws + o_wl), EMB, n); }
  { long n = (long)KP * G4P;
    repack_pad_kernel<<<(n + 255) / 256, 256, 0, stream>>>(U_lstm,
                                                        (unsigned short*)(ws + o_ubf), H_, n); }
  repack_bias_kernel<<<(G4P + 255) / 256, 256, 0, stream>>>(b_lstm, (float*)(ws + o_bp));
  { long n = (long)KP * VOCAB;
    cvt_pad_kernel<<<(long)((n + 255) / 256), 256, 0, stream>>>(W_out,
                                                        (unsigned short*)(ws + o_wo), H_, VOCAB, n); }
  // 3) xz = x @ W_lstm + b_lstm   [2048 x 2048 slab-padded], K=512
  wmma_gemm_bf16<4, false><<<dim3(G4P / 16 / 8, (B_ * TSEQ) / 64), 256, 0, stream>>>(
      (const bf16_t*)(ws + o_xbf), EMB, (const bf16_t*)(ws + o_wl), G4P,
      (const float*)(ws + o_bp), (float*)(ws + o_xz), G4P, nullptr, 0, G4P, EMB);
  // 4) all 64 LSTM steps in ONE persistent kernel (device-wide barrier per step)
  lstm_persist_kernel<<<NB_STEP, 256, 0, stream>>>(
      (const bf16_t*)(ws + o_ubf), (const float*)(ws + o_xz),
      (unsigned short*)(ws + o_h0), (unsigned short*)(ws + o_h1),
      (unsigned short*)(ws + o_hall), (int*)(ws + o_bar));
  // 5) logits = h_all @ W_out + b_out -> d_out, with fused softmax partials
  wmma_gemm_bf16<4, true><<<dim3(VOCAB / 16 / 8, (B_ * TSEQ) / 64), 256, 0, stream>>>(
      (const bf16_t*)(ws + o_hall), KP, (const bf16_t*)(ws + o_wo), VOCAB,
      b_out, out, VOCAB, (float2*)(ws + o_pst), NT_V, VOCAB, KP);
  // 6) merge partials (32 MB) then normalize in place
  stats_reduce_kernel<<<B_ * TSEQ, 256, 0, stream>>>(
      (const float2*)(ws + o_pst), (float*)(ws + o_stats));
  softmax_norm_kernel<<<dim3((VOCAB + 255) / 256, B_ * TSEQ), 256, 0, stream>>>(
      out, (float*)(ws + o_stats));
}
